// SwarmCoordination_50964081934828
// MI455X (gfx1250) — compile-verified, matrix operand order checked by reading, not run on hardware
//
#include <hip/hip_runtime.h>

#define N_NODES 2048
#define N_ANTS  2048
#define BATCHSZ 8
#define ANTS_PER_WG 16
#define WAVES_PER_WG 4
#define BLOCKS_PER_WAVE (64 / WAVES_PER_WG)   // 64 blocks of 32 nodes per row

typedef __attribute__((ext_vector_type(16))) _Float16 v16h;
typedef __attribute__((ext_vector_type(8)))  float    v8f;

__device__ __forceinline__ unsigned int pcg_hash(unsigned int x) {
    x = x * 747796405u + 2891336453u;
    unsigned int w = ((x >> ((x >> 28) + 4u)) ^ x) * 277803737u;
    return (w >> 22) ^ w;
}

// W[i][j] = pheromone^ALPHA * heuristic^BETA = p * h * h  (elementwise, HBM-bound once)
__global__ __launch_bounds__(256) void build_weights_kernel(
    const float* __restrict__ pher, const float* __restrict__ heur,
    float* __restrict__ W)
{
    int i = blockIdx.x * blockDim.x + threadIdx.x;     // over N*N/4 float4s
    float4 p = ((const float4*)pher)[i];
    float4 h = ((const float4*)heur)[i];
    float4 w;
    w.x = p.x * h.x * h.x;
    w.y = p.y * h.y * h.y;
    w.z = p.z * h.z * h.z;
    w.w = p.w * h.w * h.w;
    ((float4*)W)[i] = w;
}

// One workgroup = 16 ants, 4 waves. Each step:
//   level 1: 32-node segment sums of s=exp(logit) via v_wmma_f32_16x16x32_f16
//            (deviation d=exp(w)-1 in f16, exact +32.0 per block in f32)
//   level 2: inverse-CDF selection, one lane per ant
__global__ __launch_bounds__(WAVES_PER_WG * 32) void aco_walk_kernel(
    const float* __restrict__ W, const float* __restrict__ H,
    const int* __restrict__ pos0,
    unsigned short* __restrict__ paths, float* __restrict__ plens)
{
    __shared__ unsigned int visited[ANTS_PER_WG][64];   // 2048-bit mask per ant
    __shared__ float        bsum[ANTS_PER_WG][64];      // 32-node block sums of s
    __shared__ int          curS[ANTS_PER_WG];

    const int tid  = threadIdx.x;
    const int lane = tid & 31;
    const int wav  = tid >> 5;
    const int antL = lane & 15;      // ant this lane contributes to (A-matrix M row)
    const int hi   = lane >> 4;      // lane a / lane a+16 split the 32 K-values
    const int antBase = blockIdx.x * ANTS_PER_WG;

    for (int i = tid; i < ANTS_PER_WG * 64; i += WAVES_PER_WG * 32)
        ((unsigned int*)visited)[i] = 0u;
    __syncthreads();

    float plen = 0.0f;
    if (tid < ANTS_PER_WG) {
        int st = pos0[antBase + tid];
        curS[tid] = st;
        paths[(size_t)(antBase + tid) * N_NODES] = (unsigned short)st;
        visited[tid][st >> 5] |= (1u << (st & 31));
    }
    __syncthreads();

    v16h onesB;
    #pragma unroll
    for (int k = 0; k < 16; ++k) onesB[k] = (_Float16)1.0f;

    for (int step = 0; step < N_NODES - 1; ++step) {
        const int cur = curS[antL];
        const float* __restrict__ row = W + (size_t)cur * N_NODES;

        // ---- level 1: WMMA segment sums (each wave owns 16 of the 64 blocks)
        const int blk0 = wav * BLOCKS_PER_WAVE;
        for (int b = 0; b < BLOCKS_PER_WAVE; ++b) {
            const int blk = blk0 + b;
            const int nodeBase = blk * 32 + hi * 16;
            if (b + 1 < BLOCKS_PER_WAVE)
                __builtin_prefetch(row + nodeBase + 32, 0, 1);   // global_prefetch_b8

            unsigned int vw = visited[antL][blk];
            unsigned int vh = hi ? (vw >> 16) : (vw & 0xFFFFu);

            float4 wq[4];
            const float4* rp = (const float4*)(row + nodeBase);
            #pragma unroll
            for (int q = 0; q < 4; ++q) wq[q] = rp[q];           // 4x global_load_b128
            const float* wv = (const float*)wq;

            v16h afrag;
            #pragma unroll
            for (int k = 0; k < 16; ++k) {
                // s = visited ? exp(0)=1 : exp(w);   d = s - 1  (|d| <= ~1e-3)
                float d = ((vh >> k) & 1u) ? 0.0f : (__expf(wv[k]) - 1.0f);
                afrag[k] = (_Float16)d;
            }

            v8f acc = {};
            // D[a][j] = sum_k d[a][k]  (replicated over columns, all-ones B)
            acc = __builtin_amdgcn_wmma_f32_16x16x32_f16(
                false, afrag, false, onesB, (short)0, acc, false, false);

            // C/D layout: lane 0 col holds M=0..7 in VGPR 0..7, lane 16 holds M=8..15
            if (lane == 0) {
                #pragma unroll
                for (int r = 0; r < 8; ++r) bsum[r][blk] = 32.0f + acc[r];
            } else if (lane == 16) {
                #pragma unroll
                for (int r = 0; r < 8; ++r) bsum[8 + r][blk] = 32.0f + acc[r];
            }
        }
        __syncthreads();

        // ---- level 2: inverse-CDF draw, one lane per ant (lanes 0..15 of wave 0)
        if (tid < ANTS_PER_WG) {
            const int a = tid;
            float T = 0.0f;
            #pragma unroll 8
            for (int b = 0; b < 64; ++b) T += bsum[a][b];

            unsigned int hsh = pcg_hash(((unsigned int)step << 12) ^
                                        (unsigned int)(antBase + a) ^ 0x9E3779B9u);
            float u = (float)(hsh & 0xFFFFFFu) * (1.0f / 16777216.0f);
            float target = u * T;

            int bsel = -1; float before = 0.0f, accb = 0.0f;
            for (int b = 0; b < 64; ++b) {
                float na = accb + bsum[a][b];
                if (bsel < 0 && target < na) { bsel = b; before = accb; }
                accb = na;
            }
            if (bsel < 0) { bsel = 63; before = accb - bsum[a][63]; }
            float resid = target - before;

            unsigned int vw = visited[a][bsel];
            int nxt = -1; float accn = 0.0f;
            const float* rowp = W + (size_t)cur * N_NODES + bsel * 32;
            for (int i = 0; i < 32; ++i) {
                float s = ((vw >> i) & 1u) ? 1.0f : __expf(rowp[i]);
                accn += s;
                if (nxt < 0 && resid < accn) nxt = bsel * 32 + i;
            }
            if (nxt < 0) nxt = bsel * 32 + 31;

            plen += H[(size_t)cur * N_NODES + nxt];
            visited[a][nxt >> 5] |= (1u << (nxt & 31));
            curS[a] = nxt;
            paths[(size_t)(antBase + a) * N_NODES + step + 1] = (unsigned short)nxt;
        }
        __syncthreads();
    }

    if (tid < ANTS_PER_WG) plens[antBase + tid] = plen;
}

__global__ __launch_bounds__(256) void argmin_kernel(
    const float* __restrict__ plens, int* __restrict__ best)
{
    __shared__ float sv[256];
    __shared__ int   si[256];
    int t = threadIdx.x;
    float bv = 3.4e38f; int bi = 0;
    for (int i = t; i < N_ANTS; i += 256) {
        float v = plens[i];
        if (v < bv) { bv = v; bi = i; }
    }
    sv[t] = bv; si[t] = bi;
    __syncthreads();
    for (int s = 128; s > 0; s >>= 1) {
        if (t < s) {
            if (sv[t + s] < sv[t]) { sv[t] = sv[t + s]; si[t] = si[t + s]; }
        }
        __syncthreads();
    }
    if (t == 0) *best = si[0];
}

__global__ __launch_bounds__(256) void output_kernel(
    const unsigned short* __restrict__ paths, const int* __restrict__ best,
    float* __restrict__ out)
{
    int i = blockIdx.x * blockDim.x + threadIdx.x;   // over BATCHSZ * N_NODES
    int n = i & (N_NODES - 1);
    int b = *best;
    out[i] = (float)paths[(size_t)b * N_NODES + n];
}

extern "C" void kernel_launch(void* const* d_in, const int* in_sizes, int n_in,
                              void* d_out, int out_size, void* d_ws, size_t ws_size,
                              hipStream_t stream) {
    // inputs: 0=x [8*2048] f32 (unused beyond shape), 1=pheromone [N*N] f32,
    //         2=heuristic [N*N] f32, 3=ant_positions [2048] i32
    const float* pher = (const float*)d_in[1];
    const float* heur = (const float*)d_in[2];
    const int*   pos  = (const int*)d_in[3];

    // workspace layout (requires ~24 MB + 8 KB):
    //   [0, 16MB)        W          f32 N*N
    //   [16MB, 24MB)     paths      u16 N_ANTS*N_NODES
    //   [24MB, +8KB)     plens      f32 N_ANTS
    //   then             best index i32
    char* ws = (char*)d_ws;
    float*          Wbuf  = (float*)ws;
    unsigned short* paths = (unsigned short*)(ws + (size_t)N_NODES * N_NODES * 4);
    float*          plens = (float*)(ws + (size_t)N_NODES * N_NODES * 4
                                        + (size_t)N_ANTS * N_NODES * 2);
    int*            best  = (int*)(plens + N_ANTS);

    build_weights_kernel<<<(N_NODES * N_NODES / 4) / 256, 256, 0, stream>>>(pher, heur, Wbuf);
    aco_walk_kernel<<<N_ANTS / ANTS_PER_WG, WAVES_PER_WG * 32, 0, stream>>>(
        Wbuf, heur, pos, paths, plens);
    argmin_kernel<<<1, 256, 0, stream>>>(plens, best);
    output_kernel<<<(BATCHSZ * N_NODES) / 256, 256, 0, stream>>>(paths, best, (float*)d_out);
}
